// TransformerEncoderLayer_38070590112357
// MI455X (gfx1250) — matrix operand order repaired and checked by reading, not compiled, not run on hardware
//
#include <hip/hip_runtime.h>
#include <hip/hip_bf16.h>
#include <cstdint>
#include <cstddef>

// ---------------------------------------------------------------------------
// Types for CDNA5 WMMA (gfx1250, wave32)
// ---------------------------------------------------------------------------
typedef __bf16 bf16;
typedef __attribute__((ext_vector_type(16))) __bf16 v16bf;  // A/B fragment (16x32 / 32x16 bf16)
typedef __attribute__((ext_vector_type(8)))  __bf16 v8bf;   // 16-byte chunk
typedef __attribute__((ext_vector_type(4)))  __bf16 v4bf;   // 8-byte chunk
typedef __attribute__((ext_vector_type(8)))  float  v8f;    // C/D fragment (16x16 f32)

constexpr int Dm = 1024;   // d_model
constexpr int Hh = 16;     // heads
constexpr int Dh = 64;     // head dim
constexpr int Ff = 2048;   // d_ff
constexpr int Ss = 2048;   // seq len
constexpr int Bb = 2;      // batch
constexpr int BS = Bb * Ss;

union FragU { v16bf v; v8bf h[2]; };

__device__ __forceinline__ v8f wmma_bf16(v16bf a, v16bf b, v8f c) {
  // D(16x16,f32) = A(16x32,bf16) * B(32x16,bf16) + C
  return __builtin_amdgcn_wmma_f32_16x16x32_bf16(false, a, false, b, (short)0, c,
                                                 false, false);
}

// A-fragment: 16x32 bf16, row-major source, lda in elements.
// Layout: lanes 0-15 hold M=0..15 / K=0..15; lanes 16-31 hold
// M=0..15 / K=16..31 -> one contiguous 32B read per lane.
__device__ __forceinline__ v16bf load_a_frag(const bf16* A, int lda, int m0, int k0) {
  int lane = threadIdx.x & 31;
  int m = lane & 15, g = lane >> 4;
  return *(const v16bf*)(A + (size_t)(m0 + m) * lda + k0 + 16 * g);
}

// B-fragment: 32x16 bf16 (K x N) loaded from the TRANSPOSED operand Bt[N][K]
// (row-major, ldb in elements). Lanes 0-15 hold K=0..15 of column n=lane,
// lanes 16-31 hold K=16..31 -> one contiguous 32B read per lane.
__device__ __forceinline__ v16bf load_b_frag(const bf16* Bt, int ldb, int n0, int k0) {
  int lane = threadIdx.x & 31;
  int n = lane & 15, g = lane >> 4;
  return *(const v16bf*)(Bt + (size_t)(n0 + n) * ldb + k0 + 16 * g);
}

// C/D layout: lane holds column n = lane&15, rows r + 8*(lane>>4), r = 0..7.

// ---------------------------------------------------------------------------
// Elementwise converts
// ---------------------------------------------------------------------------
__global__ __launch_bounds__(256) void cvt_bf16_kernel(const float* __restrict__ x,
                                                       bf16* __restrict__ y, int n4) {
  int i = blockIdx.x * 256 + threadIdx.x;
  if (i < n4) {
    float4 f = ((const float4*)x)[i];
    v4bf o;
    o[0] = (bf16)f.x; o[1] = (bf16)f.y; o[2] = (bf16)f.z; o[3] = (bf16)f.w;
    ((v4bf*)y)[i] = o;
  }
}

// W[K][N] fp32 -> Wt[N][K] bf16 (LDS-tiled transpose, coalesced both sides)
__global__ __launch_bounds__(256) void transpose_cvt_kernel(const float* __restrict__ W,
                                                            bf16* __restrict__ Wt,
                                                            int K, int N) {
  __shared__ float tile[32][33];
  int bx = blockIdx.x * 32;  // N direction
  int by = blockIdx.y * 32;  // K direction
  int tx = threadIdx.x & 31;
  int ty = threadIdx.x >> 5;
#pragma unroll
  for (int i = 0; i < 32; i += 8)
    tile[ty + i][tx] = W[(size_t)(by + ty + i) * N + bx + tx];
  __syncthreads();
#pragma unroll
  for (int i = 0; i < 32; i += 8)
    Wt[(size_t)(bx + ty + i) * K + by + tx] = (bf16)tile[tx][ty + i];
}

// ---------------------------------------------------------------------------
// WMMA GEMM: C[M,N] = act( (A[M,K] @ B[K,N] + bias) * scale )
// A: bf16 row-major, Bt: bf16 [N][K] (transposed). 256 threads = 8 waves,
// workgroup tile 128x128, wave tile 32x64 (2x4 fragments), K-step 32.
// Weights/activations fit in the 192MB L2, so fragments load straight from
// global (no LDS staging) and cross-wave reuse is served by L2.
// ---------------------------------------------------------------------------
enum { OUT_BF16 = 0, OUT_F32 = 1, OUT_GELU_BF16 = 2, OUT_VT_BF16 = 3 };

__global__ __launch_bounds__(256) void gemm_bf16_wmma(
    const bf16* __restrict__ A, int lda,
    const bf16* __restrict__ Bt, int ldb,
    const float* __restrict__ bias, float scale, int mode,
    bf16* __restrict__ outb, float* __restrict__ outf, int ldc,
    int M, int N, int K) {
  (void)M; (void)N;
  int wave = threadIdx.x >> 5;
  int m0 = blockIdx.y * 128 + (wave >> 1) * 32;
  int n0 = blockIdx.x * 128 + (wave & 1) * 64;

  v8f acc[2][4] = {};
  for (int k = 0; k < K; k += 32) {
    v16bf a0 = load_a_frag(A, lda, m0, k);
    v16bf a1 = load_a_frag(A, lda, m0 + 16, k);
    v16bf b0 = load_b_frag(Bt, ldb, n0 + 0,  k);
    v16bf b1 = load_b_frag(Bt, ldb, n0 + 16, k);
    v16bf b2 = load_b_frag(Bt, ldb, n0 + 32, k);
    v16bf b3 = load_b_frag(Bt, ldb, n0 + 48, k);
    acc[0][0] = wmma_bf16(a0, b0, acc[0][0]);
    acc[0][1] = wmma_bf16(a0, b1, acc[0][1]);
    acc[0][2] = wmma_bf16(a0, b2, acc[0][2]);
    acc[0][3] = wmma_bf16(a0, b3, acc[0][3]);
    acc[1][0] = wmma_bf16(a1, b0, acc[1][0]);
    acc[1][1] = wmma_bf16(a1, b1, acc[1][1]);
    acc[1][2] = wmma_bf16(a1, b2, acc[1][2]);
    acc[1][3] = wmma_bf16(a1, b3, acc[1][3]);
  }

  int lane = threadIdx.x & 31;
  int cn = lane & 15, g = lane >> 4;
#pragma unroll
  for (int i = 0; i < 2; ++i)
#pragma unroll
    for (int t = 0; t < 4; ++t) {
      int col = n0 + t * 16 + cn;
      float bv = bias ? bias[col] : 0.0f;
#pragma unroll
      for (int r = 0; r < 8; ++r) {
        int row = m0 + i * 16 + r + 8 * g;
        float v = (acc[i][t][r] + bv) * scale;
        if (mode == OUT_GELU_BF16)
          v = 0.5f * v * (1.0f + erff(v * 0.70710678118f));  // exact gelu
        if (mode == OUT_F32) {
          outf[(size_t)row * ldc + col] = v;
        } else if (mode == OUT_VT_BF16) {
          // write V transposed: Vt[b][h][d][s]  (contiguous in s)
          int b = row >> 11, s = row & (Ss - 1);
          int h = col >> 6, d = col & (Dh - 1);
          outb[(((size_t)(b * Hh + h)) * Dh + d) * Ss + s] = (bf16)v;
        } else {
          outb[(size_t)row * ldc + col] = (bf16)v;
        }
      }
    }
}

// ---------------------------------------------------------------------------
// Flash attention (online softmax) with WMMA.
// Grid: (S/64, H, B). Block: 128 threads = 4 waves; wave owns 16 query rows.
// Per 64-key block: 8 WMMAs for S = Q*K^T (Q pre-scaled by 1/sqrt(dh)),
// online max/sum via 16-lane shuffles (row stats replicated across the
// half-group), P staged C-layout -> A-layout through per-wave LDS,
// then 8 WMMAs for O += P*V with Vt giving contiguous B-fragments.
// ---------------------------------------------------------------------------
__device__ __forceinline__ float half_group_max(float v) {
  v = fmaxf(v, __shfl_xor(v, 1, 32));
  v = fmaxf(v, __shfl_xor(v, 2, 32));
  v = fmaxf(v, __shfl_xor(v, 4, 32));
  v = fmaxf(v, __shfl_xor(v, 8, 32));
  return v;
}
__device__ __forceinline__ float half_group_sum(float v) {
  v += __shfl_xor(v, 1, 32);
  v += __shfl_xor(v, 2, 32);
  v += __shfl_xor(v, 4, 32);
  v += __shfl_xor(v, 8, 32);
  return v;
}

__global__ __launch_bounds__(128) void flash_attn_wmma(
    const bf16* __restrict__ Q,   // [B,S,D] bf16, pre-scaled by 1/8
    const bf16* __restrict__ Kc,  // [B,S,D] bf16
    const bf16* __restrict__ Vt,  // [B,H,Dh,S] bf16 (transposed)
    bf16* __restrict__ ctx) {     // [B,S,D] bf16
  __shared__ __align__(32) bf16 pbuf[4][16 * 64];
  int wave = threadIdx.x >> 5;
  int lane = threadIdx.x & 31;
  int cn = lane & 15, g = lane >> 4;
  int b = blockIdx.z, h = blockIdx.y;
  int q0 = blockIdx.x * 64 + wave * 16;

  const bf16* Qp = Q + ((size_t)b * Ss + q0) * Dm + h * Dh;
  const bf16* Kp = Kc + (size_t)b * Ss * Dm + h * Dh;
  const bf16* Vp = Vt + ((size_t)(b * Hh + h)) * Dh * Ss;

  v16bf aq0 = load_a_frag(Qp, Dm, 0, 0);   // dh = 64 -> two K-steps
  v16bf aq1 = load_a_frag(Qp, Dm, 0, 32);

  v8f o[4] = {};
  float rm[8], rl[8];
#pragma unroll
  for (int r = 0; r < 8; ++r) { rm[r] = -1e30f; rl[r] = 0.0f; }

  bf16* pw = pbuf[wave];

  for (int kb = 0; kb < Ss; kb += 64) {
    // S_ij tile (16 x 64): columns are keys kb .. kb+63
    v8f sc[4] = {};
#pragma unroll
    for (int t = 0; t < 4; ++t) {
      v16bf bk0 = load_b_frag(Kp + (size_t)kb * Dm, Dm, t * 16, 0);
      v16bf bk1 = load_b_frag(Kp + (size_t)kb * Dm, Dm, t * 16, 32);
      sc[t] = wmma_bf16(aq0, bk0, sc[t]);
      sc[t] = wmma_bf16(aq1, bk1, sc[t]);
    }
    // Online softmax per row (row r+8g, replicated across 16-lane group)
#pragma unroll
    for (int r = 0; r < 8; ++r) {
      float bm = fmaxf(fmaxf(sc[0][r], sc[1][r]), fmaxf(sc[2][r], sc[3][r]));
      bm = half_group_max(bm);
      float nm = fmaxf(rm[r], bm);
      float alpha = __expf(rm[r] - nm);
      rm[r] = nm;
      float ps = 0.0f;
#pragma unroll
      for (int t = 0; t < 4; ++t) {
        float p = __expf(sc[t][r] - nm);
        sc[t][r] = p;
        ps += p;
      }
      ps = half_group_sum(ps);
      rl[r] = rl[r] * alpha + ps;
#pragma unroll
      for (int t = 0; t < 4; ++t) o[t][r] *= alpha;
    }
    // Re-shape P: C-layout -> A-layout via per-wave LDS tile (16x64 bf16)
#pragma unroll
    for (int t = 0; t < 4; ++t)
#pragma unroll
      for (int r = 0; r < 8; ++r)
        pw[(r + 8 * g) * 64 + t * 16 + cn] = (bf16)sc[t][r];
    __syncthreads();
    v16bf ap0 = load_a_frag(pw, 64, 0, 0);
    v16bf ap1 = load_a_frag(pw, 64, 0, 32);
#pragma unroll
    for (int t = 0; t < 4; ++t) {
      v16bf bv0 = load_b_frag(Vp, Ss, t * 16, kb);
      v16bf bv1 = load_b_frag(Vp, Ss, t * 16, kb + 32);
      o[t] = wmma_bf16(ap0, bv0, o[t]);
      o[t] = wmma_bf16(ap1, bv1, o[t]);
    }
    __syncthreads();
  }
  // Normalize and store ctx (bf16, [B,S,D] with head offset)
  bf16* cp = ctx + ((size_t)b * Ss + q0) * Dm + h * Dh;
#pragma unroll
  for (int t = 0; t < 4; ++t)
#pragma unroll
    for (int r = 0; r < 8; ++r)
      cp[(size_t)(r + 8 * g) * Dm + t * 16 + cn] = (bf16)(o[t][r] / rl[r]);
}

// ---------------------------------------------------------------------------
// Residual add + LayerNorm (biased variance, eps 1e-5).
// One block per row (D=1024, 256 threads x 4 elems). bcast=1: second input
// is [S,D] broadcast over batch (the reference's attn_out[0] quirk).
// ---------------------------------------------------------------------------
__device__ __forceinline__ float wave_sum32(float v) {
#pragma unroll
  for (int m = 16; m >= 1; m >>= 1) v += __shfl_xor(v, m, 32);
  return v;
}

__global__ __launch_bounds__(256) void add_ln_kernel(
    const float* __restrict__ a, const float* __restrict__ b2, int bcast,
    const float* __restrict__ gam, const float* __restrict__ bet,
    float* __restrict__ xf, bf16* __restrict__ xb) {
  int row = blockIdx.x;
  const float* pa = a + (size_t)row * Dm;
  const float* pb = b2 + (size_t)(bcast ? (row & (Ss - 1)) : row) * Dm;
  int t = threadIdx.x;
  float v[4], sum = 0.0f, sq = 0.0f;
#pragma unroll
  for (int i = 0; i < 4; ++i) {
    int col = t + i * 256;
    float x = pa[col] + pb[col];
    v[i] = x; sum += x; sq += x * x;
  }
  sum = wave_sum32(sum); sq = wave_sum32(sq);
  __shared__ float ps[8], pq[8];
  int w = t >> 5, ln = t & 31;
  if (ln == 0) { ps[w] = sum; pq[w] = sq; }
  __syncthreads();
  if (t < 32) {
    float s1 = (ln < 8) ? ps[ln] : 0.0f;
    float s2 = (ln < 8) ? pq[ln] : 0.0f;
    s1 = wave_sum32(s1); s2 = wave_sum32(s2);
    if (ln == 0) { ps[0] = s1; pq[0] = s2; }
  }
  __syncthreads();
  float mu = ps[0] * (1.0f / Dm);
  float var = pq[0] * (1.0f / Dm) - mu * mu;
  float rs = rsqrtf(var + 1e-5f);
#pragma unroll
  for (int i = 0; i < 4; ++i) {
    int col = t + i * 256;
    float y = (v[i] - mu) * rs * gam[col] + bet[col];
    if (xf) xf[(size_t)row * Dm + col] = y;
    if (xb) xb[(size_t)row * Dm + col] = (bf16)y;
  }
}

// ---------------------------------------------------------------------------
// Host-side orchestration
// ---------------------------------------------------------------------------
extern "C" void kernel_launch(void* const* d_in, const int* in_sizes, int n_in,
                              void* d_out, int out_size, void* d_ws, size_t ws_size,
                              hipStream_t stream) {
  (void)in_sizes; (void)n_in; (void)out_size; (void)ws_size;
  const float* src    = (const float*)d_in[0];
  const float* w_q    = (const float*)d_in[1];
  const float* b_q    = (const float*)d_in[2];
  const float* w_k    = (const float*)d_in[3];
  const float* b_k    = (const float*)d_in[4];
  const float* w_v    = (const float*)d_in[5];
  const float* b_v    = (const float*)d_in[6];
  const float* w_out  = (const float*)d_in[7];
  const float* b_out  = (const float*)d_in[8];
  const float* ln1_g  = (const float*)d_in[9];
  const float* ln1_b  = (const float*)d_in[10];
  const float* ln2_g  = (const float*)d_in[11];
  const float* ln2_b  = (const float*)d_in[12];
  const float* ffn_w1 = (const float*)d_in[13];
  const float* ffn_b1 = (const float*)d_in[14];
  const float* ffn_w2 = (const float*)d_in[15];
  const float* ffn_b2 = (const float*)d_in[16];
  float* out = (float*)d_out;

  char* ws = (char*)d_ws;
  size_t off = 0;
  auto take = [&](size_t bytes) -> void* {
    void* p = ws + off;
    off = (off + bytes + 255) & ~(size_t)255;
    return p;
  };
  bf16* srcb  = (bf16*)take((size_t)BS * Dm * 2);  // src, bf16
  bf16* wqT   = (bf16*)take((size_t)Dm * Dm * 2);  // weights, transposed bf16
  bf16* wkT   = (bf16*)take((size_t)Dm * Dm * 2);
  bf16* wvT   = (bf16*)take((size_t)Dm * Dm * 2);
  bf16* woT   = (bf16*)take((size_t)Dm * Dm * 2);
  bf16* w1T   = (bf16*)take((size_t)Ff * Dm * 2);
  bf16* w2T   = (bf16*)take((size_t)Dm * Ff * 2);
  bf16* Qb    = (bf16*)take((size_t)BS * Dm * 2);  // Q (pre-scaled)
  bf16* Kb    = (bf16*)take((size_t)BS * Dm * 2);  // K
  bf16* Vtb   = (bf16*)take((size_t)BS * Dm * 2);  // V transposed [B,H,Dh,S]
  bf16* ctxb  = (bf16*)take((size_t)BS * Dm * 2);  // attention context
  float* src2 = (float*)take((size_t)Ss * Dm * 4); // attn_out[0]
  float* xf   = (float*)take((size_t)BS * Dm * 4); // LN1 output f32
  bf16* xbf   = (bf16*)take((size_t)BS * Dm * 2);  // LN1 output bf16
  bf16* h1b   = (bf16*)take((size_t)BS * Ff * 2);  // gelu(FFN1)
  float* ffnf = (float*)take((size_t)BS * Dm * 4); // FFN2 output f32

  dim3 blk256(256);

  // 1) Convert activations + weights to bf16 (weights transposed to [N][K])
  cvt_bf16_kernel<<<dim3((BS * Dm / 4 + 255) / 256), blk256, 0, stream>>>(src, srcb, BS * Dm / 4);
  transpose_cvt_kernel<<<dim3(Dm / 32, Dm / 32), blk256, 0, stream>>>(w_q, wqT, Dm, Dm);
  transpose_cvt_kernel<<<dim3(Dm / 32, Dm / 32), blk256, 0, stream>>>(w_k, wkT, Dm, Dm);
  transpose_cvt_kernel<<<dim3(Dm / 32, Dm / 32), blk256, 0, stream>>>(w_v, wvT, Dm, Dm);
  transpose_cvt_kernel<<<dim3(Dm / 32, Dm / 32), blk256, 0, stream>>>(w_out, woT, Dm, Dm);
  transpose_cvt_kernel<<<dim3(Ff / 32, Dm / 32), blk256, 0, stream>>>(ffn_w1, w1T, Dm, Ff);
  transpose_cvt_kernel<<<dim3(Dm / 32, Ff / 32), blk256, 0, stream>>>(ffn_w2, w2T, Ff, Dm);

  // 2) QKV projections (fold 1/sqrt(dh)=0.125 into Q; V written transposed)
  gemm_bf16_wmma<<<dim3(Dm / 128, BS / 128), blk256, 0, stream>>>(
      srcb, Dm, wqT, Dm, b_q, 0.125f, OUT_BF16, Qb, nullptr, Dm, BS, Dm, Dm);
  gemm_bf16_wmma<<<dim3(Dm / 128, BS / 128), blk256, 0, stream>>>(
      srcb, Dm, wkT, Dm, b_k, 1.0f, OUT_BF16, Kb, nullptr, Dm, BS, Dm, Dm);
  gemm_bf16_wmma<<<dim3(Dm / 128, BS / 128), blk256, 0, stream>>>(
      srcb, Dm, wvT, Dm, b_v, 1.0f, OUT_VT_BF16, Vtb, nullptr, Dm, BS, Dm, Dm);

  // 3) Flash attention
  flash_attn_wmma<<<dim3(Ss / 64, Hh, Bb), dim3(128), 0, stream>>>(Qb, Kb, Vtb, ctxb);

  // 4) Out projection — reference uses attn_out[0] only, so batch 0 rows
  gemm_bf16_wmma<<<dim3(Dm / 128, Ss / 128), blk256, 0, stream>>>(
      ctxb, Dm, woT, Dm, b_out, 1.0f, OUT_F32, nullptr, src2, Dm, Ss, Dm, Dm);

  // 5) x = LN1(src + src2_broadcast)
  add_ln_kernel<<<dim3(BS), blk256, 0, stream>>>(src, src2, 1, ln1_g, ln1_b, xf, xbf);

  // 6) FFN: h1 = gelu(x@W1 + b1); ffn = h1@W2 + b2
  gemm_bf16_wmma<<<dim3(Ff / 128, BS / 128), blk256, 0, stream>>>(
      xbf, Dm, w1T, Dm, ffn_b1, 1.0f, OUT_GELU_BF16, h1b, nullptr, Ff, BS, Ff, Dm);
  gemm_bf16_wmma<<<dim3(Dm / 128, BS / 128), blk256, 0, stream>>>(
      h1b, Ff, w2T, Ff, ffn_b2, 1.0f, OUT_F32, nullptr, ffnf, Dm, BS, Dm, Ff);

  // 7) out = LN2(x + ffn)
  add_ln_kernel<<<dim3(BS), blk256, 0, stream>>>(xf, ffnf, 0, ln2_g, ln2_b, out, nullptr);
}
